// proposalModel_90013924589871
// MI455X (gfx1250) — matrix operand order repaired and checked by reading, not compiled, not run on hardware
//
#include <hip/hip_runtime.h>
#include <hip/hip_bf16.h>

typedef __attribute__((ext_vector_type(16))) _Float16 v16h;
typedef __attribute__((ext_vector_type(8)))  _Float16 v8h;
typedef __attribute__((ext_vector_type(8)))  float    v8f;

// ---------------------------------------------------------------------------
// Problem constants
// ---------------------------------------------------------------------------
#define BB    8      // batch
#define NP    256    // proposals per image
#define NG    8      // gt boxes per image
#define NR    264    // rois per image (NP + NG)
#define FH    64     // feature map H/W
#define OUTSZ 7      // roi-align output
#define C0    512    // backbone channels
#define C2    2048   // conv2 out

// ---------------------------------------------------------------------------
// 1) Row-mean kernel: y[i] = mean(x[i*hw .. i*hw+hw-1])
// ---------------------------------------------------------------------------
__global__ void mean_rows_kernel(const float* __restrict__ x, float* __restrict__ y,
                                 int n_rows, int hw, float inv) {
    int i = blockIdx.x * blockDim.x + threadIdx.x;
    if (i >= n_rows) return;
    const float* p = x + (size_t)i * hw;
    float s = 0.f;
    for (int k = 0; k < hw; ++k) s += p[k];
    y[i] = s * inv;
}

// ---------------------------------------------------------------------------
// 2) IoF weights: one block per batch, one thread per proposal
// ---------------------------------------------------------------------------
__global__ __launch_bounds__(NP)
void iof_weights_kernel(const float* __restrict__ props,  // [B][NP][5]
                        const float* __restrict__ gts,    // [B][NG][4]
                        float* __restrict__ wout) {       // [B][NR]
    int b = blockIdx.x;
    int t = threadIdx.x;
    const float* p = props + (size_t)(b * NP + t) * 5;
    float x1 = p[0], y1 = p[1], x2 = p[2], y2 = p[3];
    float area = (x2 - x1 + 1.f) * (y2 - y1 + 1.f);
    float s = 0.f;
    for (int g = 0; g < NG; ++g) {
        const float* q = gts + (size_t)(b * NG + g) * 4;
        float lx = fmaxf(x1, q[0]), ly = fmaxf(y1, q[1]);
        float rx = fminf(x2, q[2]), ry = fminf(y2, q[3]);
        float ww = fmaxf(rx - lx + 1.f, 0.f);
        float hh = fmaxf(ry - ly + 1.f, 0.f);
        s += ww * hh / area;
    }
    int keep = (s >= 0.2f) ? 1 : 0;
    __shared__ int cnt;
    if (t == 0) cnt = 0;
    __syncthreads();
    if (keep) atomicAdd(&cnt, 1);
    __syncthreads();
    bool any = (cnt > 0);
    wout[b * NR + t] = any ? (float)keep : 0.f;
    if (t < NG) wout[b * NR + NP + t] = any ? 0.f : 1.f;
}

// ---------------------------------------------------------------------------
// 3) ROI-align (7x7, ratio 2, scale 1/8) fused with weighted per-image
//    accumulation.  One block per roi; 256 threads cover 512 channels.
// ---------------------------------------------------------------------------
__global__ __launch_bounds__(256)
void roi_align_acc_kernel(const float* __restrict__ feat,   // [B][C0][64][64]
                          const float* __restrict__ props,  // [B][NP][5]
                          const float* __restrict__ gts,    // [B][NG][4]
                          const float* __restrict__ wts,    // [B][NR]
                          float* __restrict__ per_img) {    // [B][C0][7][7]
    int roi = blockIdx.x;              // 0 .. B*NR-1
    int b = roi / NR, r = roi % NR;
    float wgt = wts[b * NR + r];
    if (wgt == 0.f) return;            // uniform per block

    float bx1, by1, bx2, by2;
    if (r < NP) {
        const float* p = props + (size_t)(b * NP + r) * 5;
        bx1 = p[0]; by1 = p[1]; bx2 = p[2]; by2 = p[3];
    } else {
        const float* p = gts + (size_t)(b * NG + (r - NP)) * 4;
        bx1 = p[0]; by1 = p[1]; bx2 = p[2]; by2 = p[3];
    }
    const float SC = 0.125f;
    float x1 = bx1 * SC, y1 = by1 * SC, x2 = bx2 * SC, y2 = by2 * SC;
    float rw = fmaxf(x2 - x1, 1.f), rh = fmaxf(y2 - y1, 1.f);
    float bw = rw / (float)OUTSZ, bh = rh / (float)OUTSZ;

    int t = threadIdx.x;               // channels t and t+256
    const float* fb = feat + (size_t)b * C0 * FH * FH;
    const float* f0 = fb + (size_t)t * FH * FH;
    const float* f1 = fb + (size_t)(t + 256) * FH * FH;

    for (int pos = 0; pos < OUTSZ * OUTSZ; ++pos) {
        int py = pos / OUTSZ, px = pos % OUTSZ;
        float a0 = 0.f, a1 = 0.f;
        #pragma unroll
        for (int s = 0; s < 4; ++s) {
            int iy = s >> 1, ix = s & 1;
            float yy = y1 + ((float)py + ((float)iy + 0.5f) * 0.5f) * bh;
            float xx = x1 + ((float)px + ((float)ix + 0.5f) * 0.5f) * bw;
            bool valid = (yy >= -1.f) && (yy <= (float)FH) &&
                         (xx >= -1.f) && (xx <= (float)FH);
            if (!valid) continue;
            yy = fminf(fmaxf(yy, 0.f), (float)(FH - 1));
            xx = fminf(fmaxf(xx, 0.f), (float)(FH - 1));
            int y0 = (int)floorf(yy), x0 = (int)floorf(xx);
            int y1i = min(y0 + 1, FH - 1), x1i = min(x0 + 1, FH - 1);
            float ly = yy - (float)y0, lx = xx - (float)x0;
            float hy = 1.f - ly, hx = 1.f - lx;
            float w00 = hy * hx, w01 = hy * lx, w10 = ly * hx, w11 = ly * lx;
            a0 += f0[y0 * FH + x0] * w00 + f0[y0 * FH + x1i] * w01 +
                  f0[y1i * FH + x0] * w10 + f0[y1i * FH + x1i] * w11;
            a1 += f1[y0 * FH + x0] * w00 + f1[y0 * FH + x1i] * w01 +
                  f1[y1i * FH + x0] * w10 + f1[y1i * FH + x1i] * w11;
        }
        atomicAdd(&per_img[(size_t)(b * C0 + t) * 49 + pos],        wgt * a0 * 0.25f);
        atomicAdd(&per_img[(size_t)(b * C0 + t + 256) * 49 + pos],  wgt * a1 * 0.25f);
    }
}

// ---------------------------------------------------------------------------
// 4a) im2col to f16:  A[m][k] with k = c*9 + ky*3 + kx  (matches OIHW weight
//     flattening).  Rows m >= M are zero padding.  One thread per (m, c)
//     writes 9 contiguous halves.  All the /9 %9 style math lives here,
//     *outside* the WMMA loop.
// ---------------------------------------------------------------------------
template <int CIN, int HIN, int MPAD>
__global__ __launch_bounds__(256)
void im2col_f16_kernel(const float* __restrict__ x,   // [8][CIN][HIN][HIN]
                       _Float16* __restrict__ A) {    // [MPAD][CIN*9]
    constexpr int HOUT = HIN - 2;
    constexpr int PP = HOUT * HOUT;
    constexpr int M = 8 * PP;
    constexpr int K = CIN * 9;
    int idx = blockIdx.x * blockDim.x + threadIdx.x;
    if (idx >= MPAD * CIN) return;
    int m = idx / CIN, c = idx % CIN;
    _Float16* arow = A + (size_t)m * K + c * 9;
    if (m >= M) {
        #pragma unroll
        for (int j = 0; j < 9; ++j) arow[j] = (_Float16)0.f;
        return;
    }
    int b = m / PP, p = m % PP, oy = p / HOUT, ox = p % HOUT;
    const float* src = x + ((size_t)(b * CIN + c) * HIN + oy) * HIN + ox;
    #pragma unroll
    for (int ky = 0; ky < 3; ++ky)
        #pragma unroll
        for (int kx = 0; kx < 3; ++kx)
            arow[ky * 3 + kx] = (_Float16)src[ky * HIN + kx];
}

// ---------------------------------------------------------------------------
// 4b) GEMM on WMMA f32 <= f16*f16.  D[m][n] = sum_k A[m][k] * W[n][k].
//     One wave per 16x16 tile, 4 waves/block.
//
//     CDNA5 wave32 fragment mapping (CDNA5 ISA §7.12.2):
//       A 16x32 f16 : lane row M = lane&15, elem j -> K = (j&7)+16*(j>>3)+8*hi
//         -> with row-major f16 A, that is TWO contiguous 16B chunks:
//            halves [k0+8*hi, +8) and [k0+16+8*hi, +8)   (global_load_b128 x2)
//       B 32x16 f16 : lane col N = lane&15, elem j -> K = (j&7)+16*(j>>3)+8*hi
//         -> two runs of 8 contiguous f32 weights (4 x b128) + pack-converts
//       C/D 16x16 f32: elem r -> M = r + 8*hi, N = lane&15   (hi = lane>>4)
// ---------------------------------------------------------------------------
template <int CIN, int HIN, int COUT>
__global__ __launch_bounds__(128)
void gemm_conv_wmma_kernel(const _Float16* __restrict__ A,  // [MPAD][CIN*9] f16
                           const float* __restrict__ w,     // [COUT][CIN*9]
                           const float* __restrict__ bias,  // [COUT]
                           float* __restrict__ y) {         // [8][COUT][HOUT][HOUT]
    constexpr int HOUT = HIN - 2;
    constexpr int PP = HOUT * HOUT;
    constexpr int M = 8 * PP;
    constexpr int K = CIN * 9;

    const int lane = threadIdx.x & 31;
    const int wave = threadIdx.x >> 5;
    const int hi = lane >> 4;
    const int lo = lane & 15;

    const int tile_m = blockIdx.x;
    const int tile_n = blockIdx.y * 4 + wave;
    const int n = tile_n * 16 + lo;          // B col for this lane

    const v8h*    arow  = (const v8h*)(A + (size_t)(tile_m * 16 + lo) * K);
    const float4* wrow4 = (const float4*)(w + (size_t)n * K);

    v8f acc = {};
    #pragma unroll 2
    for (int k0 = 0; k0 < K; k0 += 32) {
        // ---- A fragment: two aligned 16B loads ----
        int g = (k0 >> 3) + hi;
        v8h a0 = arow[g];
        v8h a1 = arow[g + 2];
        v16h af = __builtin_shufflevector(a0, a1,
                    0, 1, 2, 3, 4, 5, 6, 7, 8, 9, 10, 11, 12, 13, 14, 15);

        // ---- B fragment: two runs of 8 contiguous f32 -> f16 ----
        int f = (k0 >> 2) + 2 * hi;
        __builtin_prefetch(wrow4 + f + 16, 0, 1);   // next iterations' weights
        float4 b0 = wrow4[f + 0];
        float4 b1 = wrow4[f + 1];
        float4 b2 = wrow4[f + 4];
        float4 b3 = wrow4[f + 5];
        v16h bf;
        bf[0]  = (_Float16)b0.x; bf[1]  = (_Float16)b0.y;
        bf[2]  = (_Float16)b0.z; bf[3]  = (_Float16)b0.w;
        bf[4]  = (_Float16)b1.x; bf[5]  = (_Float16)b1.y;
        bf[6]  = (_Float16)b1.z; bf[7]  = (_Float16)b1.w;
        bf[8]  = (_Float16)b2.x; bf[9]  = (_Float16)b2.y;
        bf[10] = (_Float16)b2.z; bf[11] = (_Float16)b2.w;
        bf[12] = (_Float16)b3.x; bf[13] = (_Float16)b3.y;
        bf[14] = (_Float16)b3.z; bf[15] = (_Float16)b3.w;

        acc = __builtin_amdgcn_wmma_f32_16x16x32_f16(
                  false, af, false, bf, (short)0, acc, false, false);
    }

    float bv = bias[n];
    #pragma unroll
    for (int r = 0; r < 8; ++r) {
        int mm = tile_m * 16 + r + hi * 8;
        if (mm < M) {
            int bb = mm / PP, p = mm % PP;
            y[((size_t)(bb * COUT + n) * HOUT + p / HOUT) * HOUT + p % HOUT] =
                acc[r] + bv;
        }
    }
}

// ---------------------------------------------------------------------------
// 5) Final FC: out[b][o] = fc_b[o] + sum_k end_feat[b][k]*fc_w[o][k]
// ---------------------------------------------------------------------------
__global__ __launch_bounds__(320)
void fc_kernel(const float* __restrict__ gf,   // [8][2048]
               const float* __restrict__ bbm,  // [8][2048]
               const float* __restrict__ fw,   // [40][4096]
               const float* __restrict__ fb,   // [40]
               float* __restrict__ out) {      // [8][40]
    int i = blockIdx.x * blockDim.x + threadIdx.x;
    if (i >= 8 * 40) return;
    int b = i / 40, o = i % 40;
    const float* wr = fw + (size_t)o * 4096;
    const float* g = gf + (size_t)b * C2;
    const float* h = bbm + (size_t)b * C2;
    float s = fb[o];
    for (int k = 0; k < C2; ++k) s += g[k] * wr[k];
    for (int k = 0; k < C2; ++k) s += h[k] * wr[C2 + k];
    out[i] = s;
}

// ---------------------------------------------------------------------------
// launch
// ---------------------------------------------------------------------------
extern "C" void kernel_launch(void* const* d_in, const int* in_sizes, int n_in,
                              void* d_out, int out_size, void* d_ws, size_t ws_size,
                              hipStream_t stream) {
    const float* backbonefeat  = (const float*)d_in[0];  // (8,512,64,64)
    const float* backbonefeat_ = (const float*)d_in[1];  // (8,2048,16,16)
    const float* gtbboxes      = (const float*)d_in[2];  // (8,8,4)
    const float* proposals     = (const float*)d_in[3];  // (8,256,5)
    // d_in[4] = label (unused by reference output)
    const float* conv1_w = (const float*)d_in[5];        // (1024,512,3,3)
    const float* conv1_b = (const float*)d_in[6];
    const float* conv2_w = (const float*)d_in[7];        // (2048,1024,3,3)
    const float* conv2_b = (const float*)d_in[8];
    const float* fc_w    = (const float*)d_in[9];        // (40,4096)
    const float* fc_b    = (const float*)d_in[10];
    float* out = (float*)d_out;                          // (8,40)

    // workspace layout (floats, then f16; all regions 16B aligned)
    float* ws      = (float*)d_ws;
    float* gf      = ws;                          // 8*2048      = 16384
    float* bbm     = gf + 16384;                  // 8*2048      = 16384
    float* wts     = bbm + 16384;                 // 8*264       = 2112
    float* per_img = wts + 2112;                  // 8*512*49    = 200704
    float* out1    = per_img + 200704;            // 8*1024*25   = 204800
    float* out2    = out1 + 204800;               // 8*2048*9    = 147456
    _Float16* A1   = (_Float16*)(out2 + 147456);  // 208*4608 halves
    _Float16* A2   = A1 + (size_t)208 * 4608;     // 80*9216 halves

    // per_img must start at zero (atomic accumulation target)
    hipMemsetAsync(per_img, 0, (size_t)200704 * sizeof(float), stream);

    // globalfeat = mean over 16x16
    mean_rows_kernel<<<(16384 + 255) / 256, 256, 0, stream>>>(
        backbonefeat_, gf, 16384, 256, 1.f / 256.f);

    // iof -> weights
    iof_weights_kernel<<<BB, NP, 0, stream>>>(proposals, gtbboxes, wts);

    // roi-align + weighted per-image accumulate
    roi_align_acc_kernel<<<BB * NR, 256, 0, stream>>>(
        backbonefeat, proposals, gtbboxes, wts, per_img);

    // conv1: im2col (208x4608 f16) then WMMA GEMM  M=200, N=1024, K=4608
    im2col_f16_kernel<512, 7, 208><<<(208 * 512 + 255) / 256, 256, 0, stream>>>(
        per_img, A1);
    gemm_conv_wmma_kernel<512, 7, 1024><<<dim3(13, 16), 128, 0, stream>>>(
        A1, conv1_w, conv1_b, out1);

    // conv2: im2col (80x9216 f16) then WMMA GEMM  M=72, N=2048, K=9216
    im2col_f16_kernel<1024, 5, 80><<<(80 * 1024 + 255) / 256, 256, 0, stream>>>(
        out1, A2);
    gemm_conv_wmma_kernel<1024, 5, 2048><<<dim3(5, 32), 128, 0, stream>>>(
        A2, conv2_w, conv2_b, out2);

    // bbox_feat = mean over 3x3
    mean_rows_kernel<<<(16384 + 255) / 256, 256, 0, stream>>>(
        out2, bbm, 16384, 9, 1.f / 9.f);

    // fc head
    fc_kernel<<<1, 320, 0, stream>>>(gf, bbm, fc_w, fc_b, out);
}
